// EdgeFeatsConvMultNN_82798379532680
// MI455X (gfx1250) — matrix operand order repaired and checked by reading, not compiled, not run on hardware
//
#include <hip/hip_runtime.h>

#define N_NODES 100000
#define N_EDGES 1600000
#define IN_C    64
#define OUT_C   64
#define EDGE_F  16
#define BN_EPS  1e-5f

typedef __attribute__((ext_vector_type(16))) _Float16 v16h;
typedef __attribute__((ext_vector_type(8)))  _Float16 v8h;
typedef __attribute__((ext_vector_type(4)))  _Float16 v4h;
typedef __attribute__((ext_vector_type(8)))  float    v8f;
typedef __attribute__((ext_vector_type(4)))  float    vf4;

// ---------------------------------------------------------------- WMMA helpers
__device__ __forceinline__ v8f wmma16(v16h a, v16h b, v8f c) {
  // D = A(16x32 f16) * B(32x16 f16) + C(16x16 f32)
  return __builtin_amdgcn_wmma_f32_16x16x32_f16(false, a, false, b, (short)0, c,
                                                false, false);
}

// A-frag (16x32 f16) from row-major LDS tile. lane<16: K kb..kb+7,kb+16..kb+23 ;
// lane>=16: +8 on both halves. (ISA 7.12.2, 16-bit A layout)
__device__ __forceinline__ v16h load_a_frag(const _Float16* base, int stride,
                                            int m0, int kb, int lane) {
  int row = m0 + (lane & 15);
  int k   = kb + ((lane & 16) ? 8 : 0);
  const _Float16* p = base + row * stride + k;
  v8h lo = *(const v8h*)p;
  v8h hi = *(const v8h*)(p + 16);
  return __builtin_shufflevector(lo, hi, 0, 1, 2, 3, 4, 5, 6, 7, 8, 9, 10, 11,
                                 12, 13, 14, 15);
}

// B-frag (32x16 f16) from column-major (N-major) LDS weight tile:
// lane n holds B[k][n] for K kb..kb+7,kb+16..kb+23 (+8 for lanes>=16).
__device__ __forceinline__ v16h load_b_frag(const _Float16* base, int stride,
                                            int n0, int kb, int lane) {
  int row = n0 + (lane & 15);
  int k   = kb + ((lane & 16) ? 8 : 0);
  const _Float16* p = base + row * stride + k;
  v8h lo = *(const v8h*)p;
  v8h hi = *(const v8h*)(p + 16);
  return __builtin_shufflevector(lo, hi, 0, 1, 2, 3, 4, 5, 6, 7, 8, 9, 10, 11,
                                 12, 13, 14, 15);
}

// ---------------------------------------------------------------- prep kernels
__global__ void zero_kernel(float* __restrict__ p, int n) {
  int i = blockIdx.x * blockDim.x + threadIdx.x;
  if (i < n) p[i] = 0.0f;
}

// Transpose weights (K,N)->(N,Kpad) and convert f32->f16. Zero-pads We K to 96.
__global__ void prep_w_kernel(const float* __restrict__ W1,
                              const float* __restrict__ W2,
                              const float* __restrict__ We,
                              const float* __restrict__ Wr,
                              _Float16* __restrict__ W1t,
                              _Float16* __restrict__ W2t,
                              _Float16* __restrict__ Wet,
                              _Float16* __restrict__ Wrt) {
  int t = blockIdx.x * blockDim.x + threadIdx.x;
  if (t < 64 * 128) { int n = t >> 7, k = t & 127; W1t[t] = (_Float16)W1[k * 64 + n]; }
  if (t < 64 * 64)  { int n = t >> 6, k = t & 63;
                      W2t[t] = (_Float16)W2[k * 64 + n];
                      Wrt[t] = (_Float16)Wr[k * 64 + n]; }
  if (t < 64 * 96)  { int n = t / 96, k = t % 96;
                      Wet[t] = (k < 80) ? (_Float16)We[k * 64 + n] : (_Float16)0.0f; }
}

__global__ void cvt_x_kernel(const float* __restrict__ x, _Float16* __restrict__ xh,
                             int n4) {
  int i = blockIdx.x * blockDim.x + threadIdx.x;
  if (i >= n4) return;
  vf4 v = *((const vf4*)x + i);
  v4h h;
  h[0] = (_Float16)v[0]; h[1] = (_Float16)v[1];
  h[2] = (_Float16)v[2]; h[3] = (_Float16)v[3];
  *((v4h*)xh + i) = h;
}

// ---------------------------------------------------------------- edge kernel
// Persistent blocks: weights staged in LDS once, loop over 64-edge tiles.
// 256 threads = 8 waves; wave (mt-pair, nt) owns two 16x16 output tiles.
#define TILE_E      64
#define EDGE_TILES  (N_EDGES / TILE_E)   // 25000
#define EDGE_BLOCKS 2500                 // 10 tiles per block, uniform
__global__ __launch_bounds__(256) void edge_kernel(
    const _Float16* __restrict__ xh, const int* __restrict__ esrc,
    const int* __restrict__ edst, const float* __restrict__ eattr,
    const _Float16* __restrict__ W1t, const float* __restrict__ b1,
    const _Float16* __restrict__ W2t, const float* __restrict__ b2,
    const _Float16* __restrict__ Wet, const float* __restrict__ be,
    float* __restrict__ agg, float* __restrict__ cnt) {
  __shared__ _Float16 sAi[TILE_E][72];    // x_i rows (f16)
  __shared__ _Float16 sAj[TILE_E][72];    // x_j rows
  __shared__ _Float16 sAe[TILE_E][104];   // [edge_attr(16) | x_j(64) | pad0(16)]
  __shared__ _Float16 sH [TILE_E][72];    // relu(layer1) result
  __shared__ _Float16 sW1[64][136];       // W1^T  (N,Kpad)
  __shared__ _Float16 sW2[64][72];        // W2^T
  __shared__ _Float16 sWe[64][104];       // We^T (K padded to 96)
  __shared__ int   sSrc[TILE_E], sDst[TILE_E];
  __shared__ float sB1[64], sB2[64], sBe[64];

  const int tid = threadIdx.x;

  // ---- stage weights + biases ONCE per persistent block
  for (int i = tid; i < 64 * 128 / 4; i += 256) {
    int idx = i * 4, n = idx >> 7, k = idx & 127;
    *(v4h*)&sW1[n][k] = *(const v4h*)(W1t + idx);
  }
  for (int i = tid; i < 64 * 64 / 4; i += 256) {
    int idx = i * 4, n = idx >> 6, k = idx & 63;
    *(v4h*)&sW2[n][k] = *(const v4h*)(W2t + idx);
  }
  for (int i = tid; i < 64 * 96 / 4; i += 256) {
    int idx = i * 4, n = idx / 96, k = idx % 96;
    *(v4h*)&sWe[n][k] = *(const v4h*)(Wet + idx);
  }
  if (tid < 64) { sB1[tid] = b1[tid]; sB2[tid] = b2[tid]; sBe[tid] = be[tid]; }

  const int w = tid >> 5, lane = tid & 31;
  const int nt  = w & 3;
  const int mt0 = (w >> 2) * 2;
  const int col = nt * 16 + (lane & 15);
  const int rhi = (lane & 16) ? 8 : 0;

  for (int tile = blockIdx.x; tile < EDGE_TILES; tile += EDGE_BLOCKS) {
    const int e0 = tile * TILE_E;
    __syncthreads();   // protect sDst/sAe/sH reuse vs previous iteration

    if (tid < TILE_E) {
      int s = esrc[e0 + tid], d = edst[e0 + tid];
      sSrc[tid] = s; sDst[tid] = d;
      atomicAdd(&cnt[d], 1.0f);   // degree count
      // speculative prefetch of the next tile this block will process
      int nx = e0 + EDGE_BLOCKS * TILE_E;
      if (nx < N_EDGES) {
        __builtin_prefetch(&esrc[nx + tid], 0, 1);
        __builtin_prefetch(&edst[nx + tid], 0, 1);
        __builtin_prefetch(&eattr[(size_t)(nx + tid) * EDGE_F], 0, 1);
      }
    }
    __syncthreads();

    // gather x_i / x_j (16B chunks) + build edge-MLP input
    for (int i = tid; i < TILE_E * 8; i += 256) {
      int e = i >> 3, c = i & 7;
      v8h vi = *((const v8h*)(xh + sDst[e] * 64) + c);
      v8h vj = *((const v8h*)(xh + sSrc[e] * 64) + c);
      *(v8h*)&sAi[e][c * 8]      = vi;
      *(v8h*)&sAj[e][c * 8]      = vj;
      *(v8h*)&sAe[e][16 + c * 8] = vj;
    }
    {                                           // edge_attr f32 -> f16
      int e = tid >> 2, c = tid & 3;
      vf4 a = *((const vf4*)(eattr + (e0 + e) * EDGE_F) + c);
      v4h h;
      h[0] = (_Float16)a[0]; h[1] = (_Float16)a[1];
      h[2] = (_Float16)a[2]; h[3] = (_Float16)a[3];
      *(v4h*)&sAe[e][c * 4] = h;
    }
    if (tid < 128) {                            // zero K pad 80..95
      int e = tid >> 1, c = tid & 1;
      v8h z = {};
      *(v8h*)&sAe[e][80 + c * 8] = z;
    }
    __syncthreads();

    // layer 1: h = relu([x_i | x_j] @ W1 + b1), K = 128
    for (int it = 0; it < 2; ++it) {
      int mt = mt0 + it;
      v8f acc = {};
      for (int kc = 0; kc < 4; ++kc) {
        const _Float16* A = (kc < 2) ? &sAi[0][0] : &sAj[0][0];
        v16h a = load_a_frag(A, 72, mt * 16, (kc & 1) * 32, lane);
        v16h b = load_b_frag(&sW1[0][0], 136, nt * 16, kc * 32, lane);
        acc = wmma16(a, b, acc);
      }
      float bias = sB1[col];
      for (int v = 0; v < 8; ++v) {
        float hv = acc[v] + bias;
        hv = hv > 0.0f ? hv : 0.0f;
        sH[mt * 16 + rhi + v][col] = (_Float16)hv;
      }
    }
    __syncthreads();

    // layer 2 (m) + edge MLP (e) -> msg = m * relu(e) -> scatter
    for (int it = 0; it < 2; ++it) {
      int mt = mt0 + it;
      v8f am = {}, ae = {};
      for (int kc = 0; kc < 2; ++kc) {
        v16h a = load_a_frag(&sH[0][0], 72, mt * 16, kc * 32, lane);
        v16h b = load_b_frag(&sW2[0][0], 72, nt * 16, kc * 32, lane);
        am = wmma16(a, b, am);
      }
      for (int kc = 0; kc < 3; ++kc) {
        v16h a = load_a_frag(&sAe[0][0], 104, mt * 16, kc * 32, lane);
        v16h b = load_b_frag(&sWe[0][0], 104, nt * 16, kc * 32, lane);
        ae = wmma16(a, b, ae);
      }
      float b2v = sB2[col], bev = sBe[col];
      for (int v = 0; v < 8; ++v) {
        float ev = ae[v] + bev;
        ev = ev > 0.0f ? ev : 0.0f;
        float msg = (am[v] + b2v) * ev;
        int d = sDst[mt * 16 + rhi + v];
        atomicAdd(&agg[d * 64 + col], msg);
      }
    }
  }
}

// ---------------------------------------------------------------- node kernel
// out_pre = agg/max(cnt,1) + x @ W_root ; accumulate column sum / sumsq
#define TILE_N 32
__global__ __launch_bounds__(256) void node_kernel(
    const _Float16* __restrict__ xh, const _Float16* __restrict__ Wrt,
    const float* __restrict__ agg, const float* __restrict__ cnt,
    float* __restrict__ pre, float* __restrict__ stats) {
  __shared__ _Float16 sX[TILE_N][72];
  __shared__ _Float16 sW[64][72];
  __shared__ float sSum[64], sSq[64];

  const int tid = threadIdx.x;
  const int n0  = blockIdx.x * TILE_N;

  for (int i = tid; i < 64 * 64 / 4; i += 256) {
    int idx = i * 4, n = idx >> 6, k = idx & 63;
    *(v4h*)&sW[n][k] = *(const v4h*)(Wrt + idx);
  }
  {
    int r = tid >> 3, c = tid & 7;
    *(v8h*)&sX[r][c * 8] = *((const v8h*)(xh + (n0 + r) * 64) + c);
  }
  if (tid < 64) { sSum[tid] = 0.0f; sSq[tid] = 0.0f; }
  __syncthreads();

  const int w = tid >> 5, lane = tid & 31;
  const int mt = w >> 2, nt = w & 3;
  const int col = nt * 16 + (lane & 15);
  const int rhi = (lane & 16) ? 8 : 0;

  v8f acc = {};
  for (int kc = 0; kc < 2; ++kc) {
    v16h a = load_a_frag(&sX[0][0], 72, mt * 16, kc * 32, lane);
    v16h b = load_b_frag(&sW[0][0], 72, nt * 16, kc * 32, lane);
    acc = wmma16(a, b, acc);
  }

  float s = 0.0f, q = 0.0f;
  for (int v = 0; v < 8; ++v) {
    int nrow = n0 + mt * 16 + rhi + v;
    float ct = cnt[nrow];
    ct = ct > 1.0f ? ct : 1.0f;
    float val = acc[v] + agg[nrow * 64 + col] / ct;
    pre[nrow * 64 + col] = val;
    s += val;
    q += val * val;
  }
  atomicAdd(&sSum[col], s);   // LDS atomics
  atomicAdd(&sSq[col], q);
  __syncthreads();
  if (tid < 64) {
    atomicAdd(&stats[tid], sSum[tid]);
    atomicAdd(&stats[64 + tid], sSq[tid]);
  }
}

// ---------------------------------------------------------------- BN + ReLU
__global__ void bn_kernel(const float* __restrict__ pre,
                          const float* __restrict__ stats,
                          const float* __restrict__ gamma,
                          const float* __restrict__ beta,
                          float* __restrict__ out, int total, float invN) {
  int i = blockIdx.x * blockDim.x + threadIdx.x;
  if (i >= total) return;
  int c = i & 63;
  float mu  = stats[c] * invN;
  float var = stats[64 + c] * invN - mu * mu;
  float v = (pre[i] - mu) * rsqrtf(var + BN_EPS) * gamma[c] + beta[c];
  out[i] = v > 0.0f ? v : 0.0f;
}

// ---------------------------------------------------------------- launch
extern "C" void kernel_launch(void* const* d_in, const int* in_sizes, int n_in,
                              void* d_out, int out_size, void* d_ws, size_t ws_size,
                              hipStream_t stream) {
  const float* x      = (const float*)d_in[0];
  const int*   ei     = (const int*)d_in[1];      // [2, E]
  const float* eattr  = (const float*)d_in[2];
  const float* W1     = (const float*)d_in[3];
  const float* b1     = (const float*)d_in[4];
  const float* W2     = (const float*)d_in[5];
  const float* b2     = (const float*)d_in[6];
  const float* We     = (const float*)d_in[7];
  const float* be     = (const float*)d_in[8];
  const float* Wr     = (const float*)d_in[9];
  const float* gamma  = (const float*)d_in[10];
  const float* beta   = (const float*)d_in[11];
  float*       out    = (float*)d_out;

  // workspace layout (floats first -> keeps 16B alignment for f16 vectors)
  float* agg   = (float*)d_ws;           // N*64
  float* cnt   = agg + N_NODES * 64;     // N
  float* stats = cnt + N_NODES;          // 128
  float* pre   = stats + 128;            // N*64
  _Float16* xh  = (_Float16*)(pre + N_NODES * 64);  // N*64 halfs
  _Float16* W1t = xh + N_NODES * 64;     // 8192
  _Float16* W2t = W1t + 64 * 128;        // 4096
  _Float16* Wet = W2t + 64 * 64;         // 6144
  _Float16* Wrt = Wet + 64 * 96;         // 4096

  const int nzero = N_NODES * 64 + N_NODES + 128;   // agg|cnt|stats contiguous
  zero_kernel<<<(nzero + 255) / 256, 256, 0, stream>>>(agg, nzero);
  prep_w_kernel<<<(64 * 128 + 255) / 256, 256, 0, stream>>>(W1, W2, We, Wr,
                                                            W1t, W2t, Wet, Wrt);
  const int n4 = N_NODES * 64 / 4;
  cvt_x_kernel<<<(n4 + 255) / 256, 256, 0, stream>>>(x, xh, n4);

  edge_kernel<<<EDGE_BLOCKS, 256, 0, stream>>>(
      xh, ei, ei + N_EDGES, eattr, W1t, b1, W2t, b2, Wet, be, agg, cnt);

  node_kernel<<<N_NODES / TILE_N, 256, 0, stream>>>(xh, Wrt, agg, cnt, pre, stats);

  const int total = N_NODES * 64;
  bn_kernel<<<(total + 255) / 256, 256, 0, stream>>>(pre, stats, gamma, beta, out,
                                                     total, 1.0f / (float)N_NODES);
}
